// InjectedEncoder_75471165325341
// MI455X (gfx1250) — matrix operand order repaired and checked by reading, not compiled, not run on hardware
//
#include <hip/hip_runtime.h>
#include <hip/hip_bf16.h>

// ---------------------------------------------------------------------------
// InjectedEncoder for MI455X (gfx1250, wave32, WMMA).
//
// Roofline: ~47 GFLOP total but ~160MB dyn_w read 4x -> bandwidth bound.
// Strategy: pre-convert dyn_w to f16 once (80MB, L2-resident at 192MB L2),
// store perceived+normalized features directly as f16 (75MB instead of 151MB),
// and run every matmul through v_wmma_f32_16x16x32_f16 tiles.
// GEMM inner loop: BK=64 (2 WMMAs per stage), vectorized b128 global loads,
// register-staged software pipeline overlapping loads with WMMA, plus a
// global_prefetch_b8 stream prefetch two stages ahead for the B operand.
// ---------------------------------------------------------------------------

#define NF   64
#define LAT  512
#define IMG  64
#define HW   (IMG * IMG)          // 4096
#define NC   4
#define CF   9
#define FIN  (NF * CF)            // 576
#define NFREQ 6
#define FCH  (4 * NFREQ)          // 24
#define CAT  (LAT + NF * FCH)     // 2048
#define PDYN (NF * FIN + NF + NF * NF + NF + NF * FIN + NF)  // 78016
#define BATCH 16

typedef _Float16 v16h __attribute__((ext_vector_type(16)));
typedef _Float16 v8h  __attribute__((ext_vector_type(8)));
typedef _Float16 v4h  __attribute__((ext_vector_type(4)));
typedef float    v8f  __attribute__((ext_vector_type(8)));

#define AS_STRIDE 72   // halves: 144B rows (16B-aligned sub-chunks)
#define BS_STRIDE 80   // halves: 160B rows (32B-aligned v16h loads)

// ---------------------------------------------------------------------------
// Generic WMMA GEMM: C[z] = alpha * (A[z] @ B[z]) (+biasN) (+biasM) (leaky?)
//                    (+= old C if accumulate).
// A: f32 row-major MxK.  B: f32 or f16 row-major KxN.  C: f32 row-major MxN.
// Requires M % 16 == 0 and K % 64 == 0 (true for all call sites here).
// Block = 128 threads (4 waves); block tile 16(M) x 64(N) x 64(K); wave w
// owns the 16x16 tile at columns [n0 + 16w, n0 + 16w + 16).
// FULLN: N % 64 == 0 -> unpredicated vector loads + load/WMMA pipeline.
// ---------------------------------------------------------------------------
template <bool BF16, bool FULLN>
__global__ __launch_bounds__(128)
void gemm_wmma(const float* __restrict__ A, const void* __restrict__ Bv,
               float* __restrict__ C,
               int M, int N, int K, int lda, int ldb, int ldc,
               long long sA, long long sB, long long sC,
               const float* __restrict__ biasN,
               const float* __restrict__ biasM, long long sBiasM,
               float alpha, int flags)   // flags: 1 = leaky(0.2), 2 = accumulate
{
    __shared__ alignas(32) _Float16 As[16 * AS_STRIDE];
    __shared__ alignas(32) _Float16 BsT[64 * BS_STRIDE];   // stored [col][k]

    const int z = blockIdx.z;
    A += (long long)z * sA;
    C += (long long)z * sC;
    const float*    Bf = (const float*)Bv;
    const _Float16* Bh = (const _Float16*)Bv;
    const long long bOff = (long long)z * sB;

    const int tid  = threadIdx.x;
    const int wave = tid >> 5;
    const int lane = tid & 31;
    const int ln15 = lane & 15;   // A row m / C column n within tile
    const int hi   = lane >> 4;   // lane half

    const int m0 = blockIdx.y * 16;
    const int n0 = blockIdx.x * 64;

    v8f acc = {};

    if (FULLN) {
        // ---- register-staged pipeline, unpredicated vector loads --------
        float4 aR[2];
        float4 bR[8];
        v8h    bR16[4];

        auto loadStage = [&](int k0) {
            #pragma unroll
            for (int i = 0; i < 2; ++i) {
                int f4 = i * 128 + tid;            // 256 float4 of A tile
                int r = f4 >> 4, kq = f4 & 15;
                aR[i] = *(const float4*)&A[(long long)(m0 + r) * lda + k0 + kq * 4];
            }
            if (BF16) {
                #pragma unroll
                for (int i = 0; i < 4; ++i) {
                    int f8 = i * 128 + tid;        // 512 v8h of B tile
                    int r = f8 >> 3, cq = f8 & 7;
                    bR16[i] = *(const v8h*)&Bh[bOff + (long long)(k0 + r) * ldb + n0 + cq * 8];
                }
            } else {
                #pragma unroll
                for (int i = 0; i < 8; ++i) {
                    int f4 = i * 128 + tid;        // 1024 float4 of B tile
                    int r = f4 >> 4, cq = f4 & 15;
                    bR[i] = *(const float4*)&Bf[bOff + (long long)(k0 + r) * ldb + n0 + cq * 4];
                }
            }
        };
        // global_prefetch_b8 stream prefetch of the B tile two stages ahead
        auto prefetchStage = [&](int k0) {
            if (BF16) {
                if (tid < 64)   // 64 rows x 128B per row
                    __builtin_prefetch(&Bh[bOff + (long long)(k0 + tid) * ldb + n0], 0, 3);
            } else {
                int r = tid >> 1, half = tid & 1;  // 64 rows x 2 x 128B lines
                __builtin_prefetch(&Bf[bOff + (long long)(k0 + r) * ldb + n0 + half * 32], 0, 3);
            }
        };
        auto storeStage = [&]() {
            #pragma unroll
            for (int i = 0; i < 2; ++i) {
                int f4 = i * 128 + tid;
                int r = f4 >> 4, kq = f4 & 15;
                v4h h = { (_Float16)aR[i].x, (_Float16)aR[i].y,
                          (_Float16)aR[i].z, (_Float16)aR[i].w };
                *(v4h*)&As[r * AS_STRIDE + kq * 4] = h;   // one 8B LDS store
            }
            if (BF16) {
                #pragma unroll
                for (int i = 0; i < 4; ++i) {
                    int f8 = i * 128 + tid;
                    int r = f8 >> 3, c0 = (f8 & 7) * 8;
                    #pragma unroll
                    for (int j = 0; j < 8; ++j)
                        BsT[(c0 + j) * BS_STRIDE + r] = bR16[i][j];
                }
            } else {
                #pragma unroll
                for (int i = 0; i < 8; ++i) {
                    int f4 = i * 128 + tid;
                    int r = f4 >> 4, c0 = (f4 & 15) * 4;
                    BsT[(c0 + 0) * BS_STRIDE + r] = (_Float16)bR[i].x;
                    BsT[(c0 + 1) * BS_STRIDE + r] = (_Float16)bR[i].y;
                    BsT[(c0 + 2) * BS_STRIDE + r] = (_Float16)bR[i].z;
                    BsT[(c0 + 3) * BS_STRIDE + r] = (_Float16)bR[i].w;
                }
            }
        };

        loadStage(0);
        for (int k0 = 0; k0 < K; k0 += 64) {
            if (k0) __syncthreads();        // frag reads of prev stage done
            storeStage();
            __syncthreads();
            if (k0 + 64 < K) loadStage(k0 + 64);      // overlap with WMMAs
            if (k0 + 128 < K) prefetchStage(k0 + 128); // warm GL2/WGP$
            #pragma unroll
            for (int kh = 0; kh < 2; ++kh) {
                union { v16h v; v8h h[2]; } af;
                const _Float16* ar = &As[ln15 * AS_STRIDE + kh * 32 + hi * 16];
                af.h[0] = *(const v8h*)(ar);
                af.h[1] = *(const v8h*)(ar + 8);
                v16h bf = *(const v16h*)
                    &BsT[(wave * 16 + ln15) * BS_STRIDE + kh * 32 + hi * 16];
                acc = __builtin_amdgcn_wmma_f32_16x16x32_f16(
                    false, af.v, false, bf, (short)0, acc, false, false);
            }
        }
    } else {
        // ---- scalar bounds-checked path (freq GEMM, N=24, only) ---------
        for (int k0 = 0; k0 < K; k0 += 64) {
            #pragma unroll
            for (int i = 0; i < 8; ++i) {
                int flat = i * 128 + tid;          // 16x64 A tile
                int r = flat >> 6, kk = flat & 63;
                As[r * AS_STRIDE + kk] =
                    (_Float16)A[(long long)(m0 + r) * lda + k0 + kk];
            }
            #pragma unroll
            for (int i = 0; i < 32; ++i) {
                int flat = i * 128 + tid;          // 64x64 B tile
                int kk = flat >> 6, cc = flat & 63;
                float v = 0.0f;
                if (n0 + cc < N) {
                    long long idx = bOff + (long long)(k0 + kk) * ldb + (n0 + cc);
                    v = BF16 ? (float)Bh[idx] : Bf[idx];
                }
                BsT[cc * BS_STRIDE + kk] = (_Float16)v;
            }
            __syncthreads();
            #pragma unroll
            for (int kh = 0; kh < 2; ++kh) {
                union { v16h v; v8h h[2]; } af;
                const _Float16* ar = &As[ln15 * AS_STRIDE + kh * 32 + hi * 16];
                af.h[0] = *(const v8h*)(ar);
                af.h[1] = *(const v8h*)(ar + 8);
                v16h bf = *(const v16h*)
                    &BsT[(wave * 16 + ln15) * BS_STRIDE + kh * 32 + hi * 16];
                acc = __builtin_amdgcn_wmma_f32_16x16x32_f16(
                    false, af.v, false, bf, (short)0, acc, false, false);
            }
            __syncthreads();
        }
    }

    // ---- epilogue: C layout VGPR i -> m = i + 8*hi, n = ln15 ------------
    const int nn = n0 + wave * 16 + ln15;
    if (FULLN || nn < N) {
        #pragma unroll
        for (int i = 0; i < 8; ++i) {
            const int mm = m0 + i + hi * 8;
            float v = acc[i] * alpha;
            if (biasN) v += biasN[nn];
            if (biasM) v += biasM[(long long)z * sBiasM + mm];
            if (flags & 1) v = (v > 0.0f) ? v : 0.2f * v;
            const long long ci = (long long)mm * ldc + nn;
            if (flags & 2) v += C[ci];
            C[ci] = v;
        }
    }
}

// ---------------------------------------------------------------------------
// Elementwise / helper kernels
// ---------------------------------------------------------------------------
__global__ void k_cvt_f16(const float* __restrict__ in,
                          _Float16* __restrict__ out, long long n)
{
    long long i = (long long)blockIdx.x * 256 + threadIdx.x;
    if (i < n) out[i] = (_Float16)in[i];
}

__global__ void k_pe_transpose(const float* __restrict__ pe,
                               _Float16* __restrict__ peT)  // (24,4096)->(4096,24)
{
    int i = blockIdx.x * 256 + threadIdx.x;      // 24*4096
    if (i < FCH * HW) {
        int f = i / HW, p = i % HW;
        peT[p * FCH + f] = (_Float16)pe[f * HW + p];
    }
}

__global__ void k_copy_f32(const float* __restrict__ in,
                           float* __restrict__ out, int n)
{
    int i = blockIdx.x * 256 + threadIdx.x;
    if (i < n) out[i] = in[i];
}

__global__ void k_zero_f32(float* __restrict__ out, int n)
{
    int i = blockIdx.x * 256 + threadIdx.x;
    if (i < n) out[i] = 0.0f;
}

__global__ void k_copy2d(const float* __restrict__ in, float* __restrict__ out,
                         int rows, int cols, int ldi, int ldo)
{
    int i = blockIdx.x * 256 + threadIdx.x;
    if (i < rows * cols) {
        int r = i / cols, c = i % cols;
        out[r * ldo + c] = in[r * ldi + c];
    }
}

// out[b,o,p] = sum_c w_in[o,c] * x[b,c,p] + b_in[o]
__global__ void k_in_proj(const float* __restrict__ x,
                          const float* __restrict__ w,
                          const float* __restrict__ b,
                          float* __restrict__ out)
{
    long long i = (long long)blockIdx.x * 256 + threadIdx.x;
    if (i >= (long long)BATCH * NF * HW) return;
    int p   = (int)(i & (HW - 1));
    int o   = (int)((i >> 12) & (NF - 1));
    int bat = (int)(i >> 18);
    float s = b[o];
    #pragma unroll
    for (int c = 0; c < 3; ++c)
        s += w[o * 3 + c] * x[((long long)bat * 3 + c) * HW + p];
    out[i] = s;
}

// Fused perceive (identity + dilated Sobel pairs) + instance-norm, f16 out.
// One block per (batch, channel) of the 576-channel perceived tensor.
__global__ __launch_bounds__(256)
void k_perceive_inorm(const float* __restrict__ img, _Float16* __restrict__ xin)
{
    const int blk = blockIdx.x;
    const int ch  = blk % FIN;
    const int bat = blk / FIN;
    const int g   = ch / NF;            // 0 = identity, 1..8 = sobel variants
    const int c   = ch % NF;
    const float* src = img + ((long long)bat * NF + c) * HW;

    __shared__ float vals[HW];          // 16 KB
    __shared__ float rs1[256], rs2[256];

    float s1 = 0.0f, s2 = 0.0f;
    for (int t = threadIdx.x; t < HW; t += 256) {
        float v;
        if (g == 0) {
            v = src[t];
        } else {
            const int gi  = g - 1;
            const int d   = 1 << (gi >> 1);      // dilation 1,2,4,8
            const int isY = gi & 1;
            const int y = t >> 6, x = t & 63;
            float a = 0.0f;
            #pragma unroll
            for (int di = -1; di <= 1; ++di)
                #pragma unroll
                for (int dj = -1; dj <= 1; ++dj) {
                    int yy = y + di * d, xx = x + dj * d;
                    if ((unsigned)yy > 63u || (unsigned)xx > 63u) continue;
                    int grad = isY ? di : dj;     // gradient axis
                    int perp = isY ? dj : di;     // smoothing axis
                    float w = (float)grad * ((perp == 0) ? 2.0f : 1.0f) * 0.125f;
                    a += w * src[yy * 64 + xx];
                }
            v = a;
        }
        vals[t] = v;
        s1 += v;
        s2 += v * v;
    }
    rs1[threadIdx.x] = s1;
    rs2[threadIdx.x] = s2;
    __syncthreads();
    for (int off = 128; off; off >>= 1) {
        if (threadIdx.x < off) {
            rs1[threadIdx.x] += rs1[threadIdx.x + off];
            rs2[threadIdx.x] += rs2[threadIdx.x + off];
        }
        __syncthreads();
    }
    const float mean = rs1[0] * (1.0f / HW);
    const float var  = rs2[0] * (1.0f / HW) - mean * mean;
    const float inv  = rsqrtf(var + 1e-5f);
    _Float16* dst = xin + ((long long)bat * FIN + ch) * HW;
    for (int t = threadIdx.x; t < HW; t += 256)
        dst[t] = (_Float16)((vals[t] - mean) * inv);
}

// ---------------------------------------------------------------------------
// Host-side launcher
// ---------------------------------------------------------------------------
static inline void gemm(hipStream_t s, const float* A, const void* B, bool bf16,
                        float* C, int M, int N, int K,
                        int lda, int ldb, int ldc,
                        long long sA, long long sB, long long sC, int Z,
                        const float* biasN, const float* biasM,
                        long long sBiasM, float alpha, int flags)
{
    dim3 g((N + 63) / 64, M / 16, Z), b(128);
    const bool fulln = (N % 64) == 0;
    if (bf16) {
        if (fulln)
            gemm_wmma<true, true><<<g, b, 0, s>>>(A, B, C, M, N, K, lda, ldb, ldc,
                                                  sA, sB, sC, biasN, biasM, sBiasM, alpha, flags);
        else
            gemm_wmma<true, false><<<g, b, 0, s>>>(A, B, C, M, N, K, lda, ldb, ldc,
                                                   sA, sB, sC, biasN, biasM, sBiasM, alpha, flags);
    } else {
        if (fulln)
            gemm_wmma<false, true><<<g, b, 0, s>>>(A, B, C, M, N, K, lda, ldb, ldc,
                                                   sA, sB, sC, biasN, biasM, sBiasM, alpha, flags);
        else
            gemm_wmma<false, false><<<g, b, 0, s>>>(A, B, C, M, N, K, lda, ldb, ldc,
                                                    sA, sB, sC, biasN, biasM, sBiasM, alpha, flags);
    }
}

static inline size_t alignUp(size_t x) { return (x + 255) & ~(size_t)255; }

extern "C" void kernel_launch(void* const* d_in, const int* in_sizes, int n_in,
                              void* d_out, int out_size, void* d_ws, size_t ws_size,
                              hipStream_t stream)
{
    (void)in_sizes; (void)n_in; (void)out_size; (void)ws_size;
    const float* x_in   = (const float*)d_in[0];
    const float* inj_in = (const float*)d_in[1];
    const float* w_in   = (const float*)d_in[2];
    const float* b_in0  = (const float*)d_in[3];
    const float* fl_w1  = (const float*)d_in[4];
    const float* fl_b1  = (const float*)d_in[5];
    const float* fl_w2  = (const float*)d_in[6];
    const float* fl_b2  = (const float*)d_in[7];
    const float* fl_ws  = (const float*)d_in[8];
    const float* fl_bs  = (const float*)d_in[9];
    const float* dyn_w  = (const float*)d_in[10];
    const float* dyn_b  = (const float*)d_in[11];
    const float* pe     = (const float*)d_in[12];
    const float* otl_w1 = (const float*)d_in[13];
    const float* otl_b1 = (const float*)d_in[14];
    const float* otl_w2 = (const float*)d_in[15];
    const float* otl_b2 = (const float*)d_in[16];
    const float* otl_ws = (const float*)d_in[17];
    const float* otl_bs = (const float*)d_in[18];
    const float* ltl_w  = (const float*)d_in[19];
    const float* ltl_b  = (const float*)d_in[20];
    float* out_final = (float*)d_out;

    // ---- workspace carve-out (~195 MB) --------------------------------
    char* ws = (char*)d_ws;
    size_t off = 0;
    _Float16* dynw16 = (_Float16*)(ws + off); off += alignUp((size_t)LAT * PDYN * 2);
    _Float16* xin16  = (_Float16*)(ws + off); off += alignUp((size_t)BATCH * FIN * HW * 2);
    _Float16* peT16  = (_Float16*)(ws + off); off += alignUp((size_t)HW * FCH * 2);
    float* p_dyn   = (float*)(ws + off); off += alignUp((size_t)BATCH * PDYN * 4);
    float* out_img = (float*)(ws + off); off += alignUp((size_t)BATCH * NF * HW * 4);
    float* h_buf   = (float*)(ws + off); off += alignUp((size_t)BATCH * NF * HW * 4);
    float* inj0    = (float*)(ws + off); off += alignUp((size_t)BATCH * LAT * 4);
    float* inj1    = (float*)(ws + off); off += alignUp((size_t)BATCH * LAT * 4);
    float* lat     = (float*)(ws + off); off += alignUp((size_t)BATCH * LAT * 4);
    float* cat     = (float*)(ws + off); off += alignUp((size_t)BATCH * CAT * 4);
    float* t1      = (float*)(ws + off); off += alignUp((size_t)BATCH * CAT * 4);

    // ---- one-time setup ----------------------------------------------
    {
        long long n = (long long)LAT * PDYN;
        k_cvt_f16<<<dim3((unsigned)((n + 255) / 256)), 256, 0, stream>>>(dyn_w, dynw16, n);
    }
    k_pe_transpose<<<(FCH * HW + 255) / 256, 256, 0, stream>>>(pe, peT16);
    k_copy_f32<<<(BATCH * LAT + 255) / 256, 256, 0, stream>>>(inj_in, inj0, BATCH * LAT);
    k_zero_f32<<<(BATCH * LAT + 255) / 256, 256, 0, stream>>>(lat, BATCH * LAT);
    {
        long long n = (long long)BATCH * NF * HW;
        k_in_proj<<<dim3((unsigned)((n + 255) / 256)), 256, 0, stream>>>(x_in, w_in, b_in0, out_img);
    }

    float* injc = inj0;
    float* injn = inj1;

    // ---- parameter-slice offsets inside p_dyn -------------------------
    const long long o_kin  = 0;
    const long long o_bin  = o_kin + (long long)NF * FIN;       // 36864
    const long long o_kout = o_bin + NF;                        // 36928
    const long long o_bout = o_kout + (long long)NF * NF;       // 41024
    const long long o_ksk  = o_bout + NF;                       // 41088
    const long long o_bsk  = o_ksk + (long long)NF * FIN;       // 77952

    for (int c = 0; c < NC; ++c) {
        // --- latent residual MLP: injn = injc@ws+bs + leaky(injc@w1+b1)@w2+b2
        gemm(stream, injc, fl_w1, false, t1, BATCH, LAT, LAT, LAT, LAT, LAT,
             0, 0, 0, 1, fl_b1, nullptr, 0, 1.0f, 1 /*leaky*/);
        gemm(stream, injc, fl_ws, false, injn, BATCH, LAT, LAT, LAT, LAT, LAT,
             0, 0, 0, 1, fl_bs, nullptr, 0, 1.0f, 0);
        gemm(stream, t1, fl_w2, false, injn, BATCH, LAT, LAT, LAT, LAT, LAT,
             0, 0, 0, 1, fl_b2, nullptr, 0, 1.0f, 2 /*acc*/);
        { float* t = injc; injc = injn; injn = t; }

        // --- perceive + instance-norm -> xin16 (f16)
        k_perceive_inorm<<<BATCH * FIN, 256, 0, stream>>>(out_img, xin16);

        // --- dynamic parameters: p = injc @ dyn_w + dyn_b  (B in f16/L2)
        gemm(stream, injc, dynw16, true, p_dyn, BATCH, PDYN, LAT,
             LAT, PDYN, PDYN, 0, 0, 0, 1, dyn_b, nullptr, 0, 1.0f, 0);

        // --- dynamic conv (per-batch GEMMs over 4096 pixels)
        // h = leaky(k_in @ xin + b_in)
        gemm(stream, p_dyn + o_kin, xin16, true, h_buf,
             NF, HW, FIN, FIN, HW, HW,
             PDYN, (long long)FIN * HW, (long long)NF * HW, BATCH,
             nullptr, p_dyn + o_bin, PDYN, 1.0f, 1);
        // out = k_out @ h + b_out
        gemm(stream, p_dyn + o_kout, h_buf, false, out_img,
             NF, HW, NF, NF, HW, HW,
             PDYN, (long long)NF * HW, (long long)NF * HW, BATCH,
             nullptr, p_dyn + o_bout, PDYN, 1.0f, 0);
        // out += k_sk @ xin + b_sk
        gemm(stream, p_dyn + o_ksk, xin16, true, out_img,
             NF, HW, FIN, FIN, HW, HW,
             PDYN, (long long)FIN * HW, (long long)NF * HW, BATCH,
             nullptr, p_dyn + o_bsk, PDYN, 1.0f, 2);

        // --- cat = [lat, freq]; freq = (out @ pe^T) / 4096
        k_copy2d<<<(BATCH * LAT + 255) / 256, 256, 0, stream>>>(
            lat, cat, BATCH, LAT, LAT, CAT);
        gemm(stream, out_img, peT16, true, cat + LAT,
             NF, FCH, HW, HW, FCH, FCH,
             (long long)NF * HW, 0, CAT, BATCH,
             nullptr, nullptr, 0, 1.0f / (float)HW, 0);

        // --- output residual MLP (weights slice c)
        const float* w1  = otl_w1 + (long long)c * CAT * CAT;
        const float* w2  = otl_w2 + (long long)c * CAT * LAT;
        const float* wsk = otl_ws + (long long)c * CAT * LAT;
        gemm(stream, cat, w1, false, t1, BATCH, CAT, CAT, CAT, CAT, CAT,
             0, 0, 0, 1, otl_b1 + (long long)c * CAT, nullptr, 0, 1.0f, 1);
        gemm(stream, cat, wsk, false, lat, BATCH, LAT, CAT, CAT, LAT, LAT,
             0, 0, 0, 1, otl_bs + (long long)c * LAT, nullptr, 0, 1.0f, 0);
        gemm(stream, t1, w2, false, lat, BATCH, LAT, CAT, CAT, LAT, LAT,
             0, 0, 0, 1, otl_b2 + (long long)c * LAT, nullptr, 0, 1.0f, 2);
    }

    // --- final projection: out = lat @ ltl_w + ltl_b
    gemm(stream, lat, ltl_w, false, out_final, BATCH, LAT, LAT,
         LAT, LAT, LAT, 0, 0, 0, 1, ltl_b, nullptr, 0, 1.0f, 0);
}
